// MultiHeadSelfAttention2D_21157008900677
// MI455X (gfx1250) — matrix operand order repaired and checked by reading, not compiled, not run on hardware
//
#include <hip/hip_runtime.h>

typedef __attribute__((ext_vector_type(16))) _Float16 v16h;
typedef __attribute__((ext_vector_type(8)))  float    v8f;

#define B_     4
#define CIN    256
#define COUT   256
#define HEADS_ 8
#define DH     32
#define HH     48
#define WW     48
#define M_     (HH * WW)   /* 2304 */

// ---- WMMA fragment index helpers (CDNA5 ISA 7.12.2, 16-bit, wave32) ----
// A-matrix 16x32: lane&15 = row M; lanes<16 -> K {0..7,16..23},
// lanes>=16 -> K {8..15,24..31}
__device__ __forceinline__ int a_koff(int h, int lane) {
  int kb = (lane & 16) ? 8 : 0;
  return (h < 8) ? (kb + h) : (kb + 8 + h);
}
// B-matrix 32x16: lane&15 = col N; lanes<16 -> K {0..7,16..23},
// lanes>=16 -> K {8..15,24..31}
__device__ __forceinline__ int b_krow(int h, int lane) {
  int kb = (lane & 16) ? 8 : 0;
  return (h < 8) ? (kb + h) : (kb + 8 + h);
}
// C/D f32 16x16: lane&15 = col N; VGPR r -> row M = r (lanes<16) or 8+r

// ---- 2D sinusoidal positional encoding, value for channel c, pixel m ----
__device__ __forceinline__ float pe_val(int c, int m) {
  int y = m / WW;
  int x = m - y * WW;
  int cc  = (c < 128) ? c : (c - 128);
  float p = (float)((c < 128) ? y : x);
  int j = cc >> 1;                                   // arange(0,128,2)/128 = j/64
  float d = __powf(10000.0f, (float)j * (1.0f / 64.0f));
  float arg = p / d;
  return (cc & 1) ? __cosf(arg) : __sinf(arg);
}

// =====================================================================
// Kernel 1: projections.  y[o,m] = sum_c w[o,c] * x[b,c,m]  for w_v,w_z,w_q.
// Stores: vT[b,h,m,d] (f16), qT[b,h,m,d] (f16), kbuf[b,h,d,m] = z + PE (f16).
// 4 waves/block; wave -> one 16x16 (o,m) tile; 8 K-steps x 3 GEMMs = 24 WMMA.
// =====================================================================
__global__ void __launch_bounds__(128)
mhsa2d_proj_kernel(const float* __restrict__ x,
                   const float* __restrict__ wv,
                   const float* __restrict__ wz,
                   const float* __restrict__ wq,
                   _Float16* __restrict__ vT,
                   _Float16* __restrict__ kbuf,
                   _Float16* __restrict__ qT) {
  const int lane  = threadIdx.x & 31;
  const int wave  = threadIdx.x >> 5;
  const int mbase = blockIdx.x * 16;               // 144 m-tiles
  const int obase = (blockIdx.y * 4 + wave) * 16;  // 16 o-tiles
  const int bi    = blockIdx.z;

  const int orow = obase + (lane & 15);
  const int mcol = mbase + (lane & 15);

  v8f accV = {}, accZ = {}, accQ = {};

  for (int kk = 0; kk < CIN; kk += 32) {
    v16h xb, av, az, aq;
#pragma unroll
    for (int h = 0; h < 16; ++h) {
      const int kr = kk + b_krow(h, lane);
      xb[h] = (_Float16)x[(bi * CIN + kr) * M_ + mcol];
      const int ka = kk + a_koff(h, lane);
      av[h] = (_Float16)wv[orow * CIN + ka];
      az[h] = (_Float16)wz[orow * CIN + ka];
      aq[h] = (_Float16)wq[orow * CIN + ka];
    }
    accV = __builtin_amdgcn_wmma_f32_16x16x32_f16(false, av, false, xb, (short)0, accV, false, false);
    accZ = __builtin_amdgcn_wmma_f32_16x16x32_f16(false, az, false, xb, (short)0, accZ, false, false);
    accQ = __builtin_amdgcn_wmma_f32_16x16x32_f16(false, aq, false, xb, (short)0, accQ, false, false);
  }

  const int rowoff = (lane & 16) ? 8 : 0;
#pragma unroll
  for (int r = 0; r < 8; ++r) {
    const int o    = obase + rowoff + r;
    const int head = o >> 5;
    const int d    = o & 31;
    const int bh   = bi * HEADS_ + head;
    vT[(bh * M_ + mcol) * DH + d] = (_Float16)accV[r];
    qT[(bh * M_ + mcol) * DH + d] = (_Float16)accQ[r];
    kbuf[(bh * DH + d) * M_ + mcol] = (_Float16)(accZ[r] + pe_val(o, mcol));
  }
}

// =====================================================================
// Kernel 2: fused flash attention per (b, head, 16-row tile), BLOCK_N = 64.
// scores = vT(16x32) x k(32x64); online softmax amortized over 4 sub-tiles;
// out += P(16x64) x qT(64x32).  Per wave: 36 steps x 8 WMMAs = 288 WMMAs.
// =====================================================================
__global__ void __launch_bounds__(128)
mhsa2d_attn_kernel(const _Float16* __restrict__ vT,
                   const _Float16* __restrict__ kbuf,
                   const _Float16* __restrict__ qT,
                   float* __restrict__ out) {
  __shared__ _Float16 lds_p[4][16 * 64];   // per-wave P transpose tile (2 KB)

  const int lane  = threadIdx.x & 31;
  const int wave  = threadIdx.x >> 5;
  const int bi    = blockIdx.z;
  const int head  = blockIdx.y;
  const int mrow0 = (blockIdx.x * 4 + wave) * 16;   // 144 row tiles

  const int bh = bi * HEADS_ + head;
  const _Float16* __restrict__ vp = vT   + (size_t)bh * M_ * DH;
  const _Float16* __restrict__ qp = qT   + (size_t)bh * M_ * DH;
  const _Float16* __restrict__ kp = kbuf + (size_t)bh * DH * M_;

  const int col = lane & 15;

  // A fragment of v (rows mrow0..mrow0+15, K = dh), kept for the whole kernel
  v16h aV;
  {
    const int row = mrow0 + col;
#pragma unroll
    for (int h = 0; h < 16; ++h)
      aV[h] = vp[row * DH + a_koff(h, lane)];
  }

  float mstate[8], lstate[8];
#pragma unroll
  for (int r = 0; r < 8; ++r) { mstate[r] = -1e30f; lstate[r] = 0.0f; }
  v8f acc0 = {}, acc1 = {};

  const float scale = 0.17677669529663687f;   // 1/sqrt(dh)
  _Float16* const pl = lds_p[wave];
  const int rowoff = (lane & 16) ? 8 : 0;

  for (int n0 = 0; n0 < M_; n0 += 64) {
    // speculative prefetch of the next tiles (silently dropped if OOB)
    __builtin_prefetch(&kp[lane * M_ + n0 + 64]);
    __builtin_prefetch(&qp[(n0 + 64 + lane) * DH]);
    __builtin_prefetch(&qp[(n0 + 96 + lane) * DH]);

    // ---- scores: four 16x16 tiles, K = dh = 32 (one WMMA each) ----
    v16h b0, b1, b2, b3;
#pragma unroll
    for (int h = 0; h < 16; ++h) {
      const int kr = b_krow(h, lane);
      const _Float16* krow = kp + kr * M_ + n0 + col;
      b0[h] = krow[0];
      b1[h] = krow[16];
      b2[h] = krow[32];
      b3[h] = krow[48];
    }
    v8f s0 = {}, s1 = {}, s2 = {}, s3 = {};
    s0 = __builtin_amdgcn_wmma_f32_16x16x32_f16(false, aV, false, b0, (short)0, s0, false, false);
    s1 = __builtin_amdgcn_wmma_f32_16x16x32_f16(false, aV, false, b1, (short)0, s1, false, false);
    s2 = __builtin_amdgcn_wmma_f32_16x16x32_f16(false, aV, false, b2, (short)0, s2, false, false);
    s3 = __builtin_amdgcn_wmma_f32_16x16x32_f16(false, aV, false, b3, (short)0, s3, false, false);

    // ---- online softmax: reductions amortized over 4 sub-tiles ----
#pragma unroll
    for (int r = 0; r < 8; ++r) {
      const float x0 = s0[r] * scale, x1 = s1[r] * scale;
      const float x2 = s2[r] * scale, x3 = s3[r] * scale;
      float tm = fmaxf(fmaxf(x0, x1), fmaxf(x2, x3));
      tm = fmaxf(tm, __shfl_xor(tm, 1, 32));
      tm = fmaxf(tm, __shfl_xor(tm, 2, 32));
      tm = fmaxf(tm, __shfl_xor(tm, 4, 32));
      tm = fmaxf(tm, __shfl_xor(tm, 8, 32));
      const float mn = fmaxf(mstate[r], tm);
      const float al = __expf(mstate[r] - mn);
      const float p0 = __expf(x0 - mn);
      const float p1 = __expf(x1 - mn);
      const float p2 = __expf(x2 - mn);
      const float p3 = __expf(x3 - mn);
      float rs = (p0 + p1) + (p2 + p3);
      rs += __shfl_xor(rs, 1, 32);
      rs += __shfl_xor(rs, 2, 32);
      rs += __shfl_xor(rs, 4, 32);
      rs += __shfl_xor(rs, 8, 32);
      lstate[r] = lstate[r] * al + rs;
      mstate[r] = mn;
      s0[r] = p0; s1[r] = p1; s2[r] = p2; s3[r] = p3;
      acc0[r] *= al; acc1[r] *= al;
    }

    // ---- D-layout -> A-layout transpose of P through per-wave LDS ----
#pragma unroll
    for (int r = 0; r < 8; ++r) {
      _Float16* prow = pl + (rowoff + r) * 64 + col;
      prow[0]  = (_Float16)s0[r];
      prow[16] = (_Float16)s1[r];
      prow[32] = (_Float16)s2[r];
      prow[48] = (_Float16)s3[r];
    }
    asm volatile("s_wait_dscnt 0" ::: "memory");   // LDS in-order per wave; fence compiler too

    // ---- out += P x qT : two A fragments (K=n0.., K=n0+32..), 4 WMMAs ----
    v16h ap0, ap1, q0, q1, q2, q3;
#pragma unroll
    for (int h = 0; h < 16; ++h) {
      const int ko = a_koff(h, lane);
      ap0[h] = pl[col * 64 + ko];
      ap1[h] = pl[col * 64 + 32 + ko];
      const int kr = n0 + b_krow(h, lane);
      q0[h] = qp[kr * DH + col];
      q1[h] = qp[kr * DH + 16 + col];
      q2[h] = qp[(kr + 32) * DH + col];
      q3[h] = qp[(kr + 32) * DH + 16 + col];
    }
    acc0 = __builtin_amdgcn_wmma_f32_16x16x32_f16(false, ap0, false, q0, (short)0, acc0, false, false);
    acc1 = __builtin_amdgcn_wmma_f32_16x16x32_f16(false, ap0, false, q1, (short)0, acc1, false, false);
    acc0 = __builtin_amdgcn_wmma_f32_16x16x32_f16(false, ap1, false, q2, (short)0, acc0, false, false);
    acc1 = __builtin_amdgcn_wmma_f32_16x16x32_f16(false, ap1, false, q3, (short)0, acc1, false, false);
  }

  // ---- epilogue: normalize and scatter to (b, c_out, h, w) ----
#pragma unroll
  for (int r = 0; r < 8; ++r) {
    const int m = mrow0 + rowoff + r;
    const float inv = 1.0f / lstate[r];
    out[(bi * COUT + head * DH + col)      * M_ + m] = acc0[r] * inv;
    out[(bi * COUT + head * DH + 16 + col) * M_ + m] = acc1[r] * inv;
  }
}

// =====================================================================
extern "C" void kernel_launch(void* const* d_in, const int* in_sizes, int n_in,
                              void* d_out, int out_size, void* d_ws, size_t ws_size,
                              hipStream_t stream) {
  const float* x  = (const float*)d_in[0];
  const float* wv = (const float*)d_in[1];
  const float* wz = (const float*)d_in[2];
  const float* wq = (const float*)d_in[3];
  float* out = (float*)d_out;

  const size_t nElem = (size_t)B_ * HEADS_ * M_ * DH;   // 2,359,296 f16 each
  _Float16* vT   = (_Float16*)d_ws;
  _Float16* kbuf = vT + nElem;
  _Float16* qT   = kbuf + nElem;

  dim3 pgrid(M_ / 16, 4, B_);            // 144 m-tiles x 4 o-groups x batch
  mhsa2d_proj_kernel<<<pgrid, 128, 0, stream>>>(x, wv, wz, wq, vT, kbuf, qT);

  dim3 agrid(M_ / 64, HEADS_, B_);       // 36 row-blocks x heads x batch
  mhsa2d_attn_kernel<<<agrid, 128, 0, stream>>>(vT, kbuf, qT, out);
}
